// RGCNEncoder_62508954026233
// MI455X (gfx1250) — matrix operand order repaired and checked by reading, not compiled, not run on hardware
//
#include <hip/hip_runtime.h>

#define ND    20000
#define NPROT 40000
#define NDIS  10000
#define NTOT  (ND + NPROT + NDIS)       // 70000
#define RREL  4
#define BB    4
#define HDIM  128
#define ODIM  64
#define EREL  400000
#define KTOT  (RREL * HDIM + HDIM)      // 640: [means(512) | x(128)]
#define NKSTEP (KTOT / 32)              // 20 WMMA K-steps

typedef __attribute__((ext_vector_type(16))) __bf16 v16bf;
typedef __attribute__((ext_vector_type(8)))  __bf16 v8bf;
typedef __attribute__((ext_vector_type(2)))  __bf16 v2bf;
typedef __attribute__((ext_vector_type(8)))  float  v8f;

// ---------------- zero fill ----------------
__global__ void k_zero(float* __restrict__ p, long n) {
    long i = (long)blockIdx.x * blockDim.x + threadIdx.x;
    long st = (long)gridDim.x * blockDim.x;
    for (; i < n; i += st) p[i] = 0.0f;
}

// ---------------- gather initial node features ----------------
__global__ void k_gather(const int* __restrict__ xd, const int* __restrict__ xp,
                         const int* __restrict__ xs,
                         const float* __restrict__ ed, const float* __restrict__ ep,
                         const float* __restrict__ es, float* __restrict__ h) {
    long i = (long)blockIdx.x * blockDim.x + threadIdx.x;
    long n = (long)NTOT * HDIM;
    long st = (long)gridDim.x * blockDim.x;
    for (; i < n; i += st) {
        int node = (int)(i / HDIM), f = (int)(i % HDIM);
        float v;
        if (node < ND)              v = ed[(long)xd[node] * HDIM + f];
        else if (node < ND + NPROT) v = ep[(long)xp[node - ND] * HDIM + f];
        else                        v = es[(long)xs[node - ND - NPROT] * HDIM + f];
        h[i] = v;
    }
}

// ---------------- per-edge scatter-add: one wave32 per edge ----------------
__global__ void k_scatter(const int* __restrict__ dp, const int* __restrict__ pp,
                          const int* __restrict__ pd, const int* __restrict__ dd,
                          const float* __restrict__ x,
                          float* __restrict__ sums, float* __restrict__ cnt) {
    long tid = (long)blockIdx.x * blockDim.x + threadIdx.x;
    long w   = tid >> 5;
    int lane = (int)(tid & 31);
    long nw  = (long)RREL * EREL;
    long wst = ((long)gridDim.x * blockDim.x) >> 5;
    for (; w < nw; w += wst) {
        int r = (int)(w / EREL);
        int e = (int)(w % EREL);
        int s, d;
        if (r == 0)      { s = dp[e];          d = dp[EREL + e] + ND; }
        else if (r == 1) { s = pp[e] + ND;     d = pp[EREL + e] + ND; }
        else if (r == 2) { s = pd[e] + ND;     d = pd[EREL + e] + ND + NPROT; }
        else             { s = dd[e];          d = dd[EREL + e] + ND + NPROT; }
        long seg = (long)d * RREL + r;
        const float* xs = x + (long)s * HDIM;
        float* ss = sums + seg * HDIM;
        #pragma unroll
        for (int j = 0; j < HDIM / 32; ++j)
            atomicAdd(&ss[lane + j * 32], xs[lane + j * 32]);
        if (lane == 0) atomicAdd(&cnt[seg], 1.0f);
    }
}

// ---------------- build bf16 Wbig[KTOT, out] = [comp@basis ; root] ----------
__global__ void k_buildw(const float* __restrict__ comp, const float* __restrict__ basis,
                         const float* __restrict__ root, __bf16* __restrict__ W, int out) {
    long i = (long)blockIdx.x * blockDim.x + threadIdx.x;
    long n = (long)KTOT * out;
    long st = (long)gridDim.x * blockDim.x;
    for (; i < n; i += st) {
        int k = (int)(i / out), o = (int)(i % out);
        float v;
        if (k < RREL * HDIM) {
            int r = k / HDIM, ii = k % HDIM;
            v = 0.0f;
            #pragma unroll
            for (int b = 0; b < BB; ++b)
                v += comp[r * BB + b] * basis[((long)b * HDIM + ii) * out + o];
        } else {
            v = root[(long)(k - RREL * HDIM) * out + o];
        }
        W[i] = (__bf16)v;
    }
}

// ------- pack W into fragment-major layout: Wp[ct][s][lane][16] bf16 --------
// Lane's B fragment for k-step s = W[s*32+half*8 .. +8) then W[s*32+16+half*8 .. +8), [col]
__global__ void k_packw(const __bf16* __restrict__ W, __bf16* __restrict__ Wp, int out) {
    long n = (long)(out / 16) * NKSTEP * 32 * 16;
    long i = (long)blockIdx.x * blockDim.x + threadIdx.x;
    long st = (long)gridDim.x * blockDim.x;
    for (; i < n; i += st) {
        int j = (int)(i & 15);
        long t = i >> 4;
        int l = (int)(t & 31); t >>= 5;
        int s = (int)(t % NKSTEP);
        int ct = (int)(t / NKSTEP);
        int k   = s * 32 + (l >> 4) * 8 + (j & 7) + (j >> 3) * 16;
        int col = ct * 16 + (l & 15);
        Wp[i] = W[(long)k * out + col];
    }
}

// ---------------- build bf16 A[N, 640] = [means | x], 32-bit stores ---------
__global__ void k_builda(const float* __restrict__ x, const float* __restrict__ sums,
                         const float* __restrict__ cnt, __bf16* __restrict__ A) {
    long n = (long)NTOT * (KTOT / 2);
    long i = (long)blockIdx.x * blockDim.x + threadIdx.x;
    long st = (long)gridDim.x * blockDim.x;
    for (; i < n; i += st) {
        int node = (int)(i / (KTOT / 2));
        int kp   = (int)(i % (KTOT / 2)) * 2;       // even k; pair never crosses regions
        float v0, v1;
        if (kp < RREL * HDIM) {
            int r = kp >> 7, hh = kp & 127;
            long seg = (long)node * RREL + r;
            float inv = 1.0f / fmaxf(cnt[seg], 1.0f);
            const float* sp = sums + seg * HDIM + hh;
            v0 = sp[0] * inv;
            v1 = sp[1] * inv;
        } else {
            const float* xp = x + (long)node * HDIM + (kp - RREL * HDIM);
            v0 = xp[0];
            v1 = xp[1];
        }
        v2bf pr; pr[0] = (__bf16)v0; pr[1] = (__bf16)v1;
        *(v2bf*)(A + (long)node * KTOT + kp) = pr;
    }
}

// ---------------- WMMA GEMM: y[N,OUT] = A[N,640] x Wp + bias ----------------
// One wave per 16x32 output slab (two 16x16 col tiles): A fragment loaded once,
// two packed B fragments, two back-to-back WMMAs (independent accumulators).
// EXEC all-ones (70000 = 16*4375 exact tiling).
template <int OUT>
__global__ void __launch_bounds__(128)
k_gemm(const __bf16* __restrict__ A, const __bf16* __restrict__ Wp,
       const float* __restrict__ bias, float* __restrict__ y) {
    const int lane     = threadIdx.x;          // 0..31
    const int m        = lane & 15;
    const int half     = lane >> 4;
    const int pairIdx  = blockIdx.x * blockDim.y + threadIdx.y;
    const int colTile0 = pairIdx * 2;
    const int rowBase  = blockIdx.y * 16;

    union FragA { v16bf v; v8bf h[2]; };

    // per-lane base pointers (16B aligned: row*1280 + half*16 bytes)
    const __bf16* Arow   = A + (long)(rowBase + m) * KTOT + half * 8;
    const __bf16* Wlane0 = Wp + ((long)colTile0 * NKSTEP * 32 + lane) * 16;
    const __bf16* Wlane1 = Wlane0 + (long)NKSTEP * 32 * 16;   // next col tile block

    v8f acc0 = {};
    v8f acc1 = {};
    #pragma unroll 2
    for (int s = 0; s < NKSTEP; ++s) {
        FragA af;
        af.h[0] = *(const v8bf*)(Arow + s * 32);        // K = s*32+half*8 .. +8
        af.h[1] = *(const v8bf*)(Arow + s * 32 + 16);   // K = s*32+half*8+16 .. +8
        v16bf bv0 = *(const v16bf*)(Wlane0 + (long)s * 32 * 16);
        v16bf bv1 = *(const v16bf*)(Wlane1 + (long)s * 32 * 16);
        if (s + 2 < NKSTEP)
            __builtin_prefetch(Arow + (s + 2) * 32, 0, 3);
        acc0 = __builtin_amdgcn_wmma_f32_16x16x32_bf16(
            false, af.v, false, bv0, (short)0, acc0, false, false);
        acc1 = __builtin_amdgcn_wmma_f32_16x16x32_bf16(
            false, af.v, false, bv1, (short)0, acc1, false, false);
    }
    #pragma unroll
    for (int v = 0; v < 8; ++v) {
        int row  = rowBase + half * 8 + v;
        int col0 = colTile0 * 16 + m;
        int col1 = col0 + 16;
        y[(long)row * OUT + col0] = acc0[v] + bias[col0];
        y[(long)row * OUT + col1] = acc1[v] + bias[col1];
    }
}

// ---------------- per-type LayerNorm (+optional ReLU), wave per row ---------
__global__ void k_ln(const float* __restrict__ y, const float* __restrict__ g,
                     const float* __restrict__ b, float* __restrict__ outp,
                     int out, int relu) {
    int wpb  = blockDim.x >> 5;
    long row = (long)blockIdx.x * wpb + (threadIdx.x >> 5);
    int lane = threadIdx.x & 31;
    if (row >= NTOT) return;
    int t = (row < ND) ? 0 : ((row < ND + NPROT) ? 1 : 2);
    const float* yr = y + row * out;
    int nc = out >> 5;                       // <= 4
    float vals[4];
    float s = 0.0f, s2 = 0.0f;
    for (int ci = 0; ci < nc; ++ci) {
        float v = yr[lane + ci * 32];
        vals[ci] = v; s += v; s2 += v * v;
    }
    #pragma unroll
    for (int off = 16; off >= 1; off >>= 1) {
        s  += __shfl_xor(s,  off, 32);
        s2 += __shfl_xor(s2, off, 32);
    }
    float mu  = s / out;
    float var = s2 / out - mu * mu;
    float inv = rsqrtf(var + 1e-5f);
    for (int ci = 0; ci < nc; ++ci) {
        int c = lane + ci * 32;
        float v = (vals[ci] - mu) * inv * g[t * out + c] + b[t * out + c];
        if (relu) v = fmaxf(v, 0.0f);
        outp[row * out + c] = v;
    }
}

// ---------------- host side ----------------
static inline size_t alignup(size_t x) { return (x + 255) & ~(size_t)255; }

extern "C" void kernel_launch(void* const* d_in, const int* in_sizes, int n_in,
                              void* d_out, int out_size, void* d_ws, size_t ws_size,
                              hipStream_t stream) {
    const int*   x_drug  = (const int*)d_in[0];
    const int*   x_prot  = (const int*)d_in[1];
    const int*   x_dis   = (const int*)d_in[2];
    const int*   ei_dp   = (const int*)d_in[3];
    const int*   ei_pp   = (const int*)d_in[4];
    const int*   ei_pd   = (const int*)d_in[5];
    const int*   ei_dd   = (const int*)d_in[6];
    const float* emb_d   = (const float*)d_in[7];
    const float* emb_p   = (const float*)d_in[8];
    const float* emb_s   = (const float*)d_in[9];
    const float* comp0   = (const float*)d_in[10];
    const float* basis0  = (const float*)d_in[11];
    const float* root0   = (const float*)d_in[12];
    const float* bias0   = (const float*)d_in[13];
    const float* comp1   = (const float*)d_in[14];
    const float* basis1  = (const float*)d_in[15];
    const float* root1   = (const float*)d_in[16];
    const float* bias1   = (const float*)d_in[17];
    const float* ln0_g   = (const float*)d_in[18];
    const float* ln0_b   = (const float*)d_in[19];
    const float* ln1_g   = (const float*)d_in[20];
    const float* ln1_b   = (const float*)d_in[21];

    // workspace carving
    char* ws = (char*)d_ws;
    size_t off = 0;
    float*  sums = (float*)(ws + off); off += alignup((size_t)NTOT * RREL * HDIM * 4);
    float*  cnt  = (float*)(ws + off); off += alignup((size_t)NTOT * RREL * 4);
    float*  hx   = (float*)(ws + off); off += alignup((size_t)NTOT * HDIM * 4);
    float*  yv   = (float*)(ws + off); off += alignup((size_t)NTOT * HDIM * 4);
    __bf16* Abf  = (__bf16*)(ws + off); off += alignup((size_t)NTOT * KTOT * 2);
    __bf16* W0   = (__bf16*)(ws + off); off += alignup((size_t)KTOT * HDIM * 2);
    __bf16* W1   = (__bf16*)(ws + off); off += alignup((size_t)KTOT * ODIM * 2);
    __bf16* Wp0  = (__bf16*)(ws + off); off += alignup((size_t)KTOT * HDIM * 2);
    __bf16* Wp1  = (__bf16*)(ws + off); off += alignup((size_t)KTOT * ODIM * 2);
    (void)ws_size; (void)n_in; (void)in_sizes; (void)out_size;

    const int TPB = 256;
    long nW0 = (long)KTOT * HDIM, nW1 = (long)KTOT * ODIM;
    long nH  = (long)NTOT * HDIM;
    long nS  = (long)NTOT * RREL * HDIM;
    long nC  = (long)NTOT * RREL;
    long nA2 = (long)NTOT * (KTOT / 2);
    long nEdgeThreads = (long)RREL * EREL * 32;

    // weights (both layers) + initial features
    k_buildw<<<dim3((unsigned)((nW0 + TPB - 1) / TPB)), TPB, 0, stream>>>(comp0, basis0, root0, W0, HDIM);
    k_buildw<<<dim3((unsigned)((nW1 + TPB - 1) / TPB)), TPB, 0, stream>>>(comp1, basis1, root1, W1, ODIM);
    k_packw<<<dim3((unsigned)((nW0 + TPB - 1) / TPB)), TPB, 0, stream>>>(W0, Wp0, HDIM);
    k_packw<<<dim3((unsigned)((nW1 + TPB - 1) / TPB)), TPB, 0, stream>>>(W1, Wp1, ODIM);
    k_gather<<<dim3((unsigned)((nH + TPB - 1) / TPB)), TPB, 0, stream>>>(x_drug, x_prot, x_dis,
                                                                          emb_d, emb_p, emb_s, hx);

    // ---- layer 0 ----
    k_zero<<<dim3((unsigned)((nS + TPB - 1) / TPB)), TPB, 0, stream>>>(sums, nS);
    k_zero<<<dim3((unsigned)((nC + TPB - 1) / TPB)), TPB, 0, stream>>>(cnt, nC);
    k_scatter<<<dim3((unsigned)((nEdgeThreads + TPB - 1) / TPB)), TPB, 0, stream>>>(
        ei_dp, ei_pp, ei_pd, ei_dd, hx, sums, cnt);
    k_builda<<<dim3((unsigned)((nA2 + TPB - 1) / TPB)), TPB, 0, stream>>>(hx, sums, cnt, Abf);
    // OUT=128: 8 col tiles = 4 pairs -> block(32,4), grid(1, 4375)
    k_gemm<HDIM><<<dim3(1, NTOT / 16), dim3(32, 4), 0, stream>>>(Abf, Wp0, bias0, yv);
    k_ln<<<dim3(NTOT / 8), 256, 0, stream>>>(yv, ln0_g, ln0_b, hx, HDIM, 1);

    // ---- layer 1 ----
    k_zero<<<dim3((unsigned)((nS + TPB - 1) / TPB)), TPB, 0, stream>>>(sums, nS);
    k_zero<<<dim3((unsigned)((nC + TPB - 1) / TPB)), TPB, 0, stream>>>(cnt, nC);
    k_scatter<<<dim3((unsigned)((nEdgeThreads + TPB - 1) / TPB)), TPB, 0, stream>>>(
        ei_dp, ei_pp, ei_pd, ei_dd, hx, sums, cnt);
    k_builda<<<dim3((unsigned)((nA2 + TPB - 1) / TPB)), TPB, 0, stream>>>(hx, sums, cnt, Abf);
    // OUT=64: 4 col tiles = 2 pairs -> block(32,2), grid(1, 4375)
    k_gemm<ODIM><<<dim3(1, NTOT / 16), dim3(32, 2), 0, stream>>>(Abf, Wp1, bias1, yv);
    k_ln<<<dim3(NTOT / 8), 256, 0, stream>>>(yv, ln1_g, ln1_b, (float*)d_out, ODIM, 0);
}
